// SREXmodel_5772436046131
// MI455X (gfx1250) — matrix operand order repaired and checked by reading, not compiled, not run on hardware
//
#include <hip/hip_runtime.h>

// ---------------------------------------------------------------------------
// CDNA5 / gfx1250 implementation of the SREX GNN reference.
// Heavy linears -> v_wmma_f32_16x16x32_f16 (f16 in, f32 accumulate),
// weights pre-packed into WMMA fragment layout; each wave computes a 64x32
// output strip (4 M-tiles x 2 N-tiles): 8 WMMAs per 32-wide K step on
// 12 b128 loads, A fragments reused across the two N accumulators.
// Irregular GAT edge softmax / BN stats / route cumsums -> f32 VALU + atomics.
// ---------------------------------------------------------------------------

typedef __attribute__((ext_vector_type(16))) _Float16 v16h;
typedef __attribute__((ext_vector_type(8)))  float    v8f;

#define LEAKY01(v) ((v) > 0.f ? (v) : 0.01f * (v))

__device__ __forceinline__ void atomicMaxFloat(float* addr, float value) {
  // IEEE trick: positive floats order as ints, negative floats reverse-order as uints
  if (value >= 0.f) atomicMax((int*)addr, __float_as_int(value));
  else              atomicMin((unsigned int*)addr, __float_as_uint(value));
}

union Frag { uint4 q[2]; v16h v; };

// ---------------------------------------------------------------------------
// Weight packer: W f32 [Ktrue, N] row-major -> f16 fragment-major, K padded
// to Kt (multiple of 32, zero fill). Tile (kt, nt) is 512 contiguous halfs:
//   off = ((kt*ntiles + nt)*32 + lane)*16 + e,  k = kt*32 + e + 16*(lane>=16),
//   n = nt*16 + (lane&15)   -- contiguous k order, same as the A loads in gemm_wmma_k.
// Adjacent nt tiles are contiguous -> 2-tile B fragments are 2KB streams.
// ---------------------------------------------------------------------------
__global__ void pack_w_k(const float* __restrict__ W, _Float16* __restrict__ out,
                         int Ktrue, int Kt, int N)
{
  int idx = blockIdx.x * blockDim.x + threadIdx.x;
  if (idx >= Kt * N) return;
  int e = idx & 15;
  int l = (idx >> 4) & 31;
  int t = idx >> 9;
  int ntiles = N >> 4;
  int nt = t % ntiles;
  int kt = t / ntiles;
  int k = kt * 32 + e + ((l >> 4) << 4);
  int n = nt * 16 + (l & 15);
  out[idx] = (k < Ktrue) ? (_Float16)W[(size_t)k * N + n] : (_Float16)0.f;
}

// pad [rows,16] f32 -> [rows,32] f16 (zero cols 16..31) for the K=16 linears
__global__ void pad_x_f16_k(const float* __restrict__ in, _Float16* __restrict__ out, int rows)
{
  int idx = blockIdx.x * blockDim.x + threadIdx.x;
  if (idx >= rows * 32) return;
  int c = idx & 31, r = idx >> 5;
  out[idx] = (c < 16) ? (_Float16)in[r * 16 + c] : (_Float16)0.f;
}

// ---------------------------------------------------------------------------
// WMMA GEMM: C[M,N] = act(A[M,Kt](f16, row-major, zero-padded K) @ B + bias)
// B pre-packed by pack_w_k. Wave: 64x32 strip (4 M-tiles x 2 N-tiles).
// Block = 4 waves stacked in M = 256x32. Requires M%256==0, N%32==0, Kt%32==0.
// ---------------------------------------------------------------------------
__global__ void __launch_bounds__(128)
gemm_wmma_k(const _Float16* __restrict__ A, const _Float16* __restrict__ Bp,
            const float* __restrict__ bias, float* __restrict__ outF,
            _Float16* __restrict__ outH, int M, int N, int Kt, int act)
{
  const int lane  = threadIdx.x & 31;
  const int wave  = threadIdx.x >> 5;
  const int nt0   = blockIdx.x * 2;              // two adjacent N-tiles per wave
  const int hs    = lane >> 4;
  const int lm    = lane & 15;
  const int row0  = blockIdx.y * 256 + wave * 64;
  const int klo   = hs * 16;
  const int ntiles = N >> 4;

  const _Float16* Ar0 = A + (size_t)(row0 + lm) * Kt;
  const _Float16* Bl  = Bp + ((size_t)nt0 * 32 + lane) * 16;
  const size_t bstep  = (size_t)ntiles * 512;    // halfs per K-tile of packed B

  v8f acc[4][2];
#pragma unroll
  for (int i = 0; i < 4; ++i) {
    acc[i][0] = (v8f){0.f,0.f,0.f,0.f,0.f,0.f,0.f,0.f};
    acc[i][1] = (v8f){0.f,0.f,0.f,0.f,0.f,0.f,0.f,0.f};
  }

  for (int k0 = 0; k0 < Kt; k0 += 32) {
    const _Float16* bp = Bl + (size_t)(k0 >> 5) * bstep;
    Frag fb0, fb1;
    fb0.q[0] = *reinterpret_cast<const uint4*>(bp);
    fb0.q[1] = *reinterpret_cast<const uint4*>(bp + 8);
    fb1.q[0] = *reinterpret_cast<const uint4*>(bp + 512);
    fb1.q[1] = *reinterpret_cast<const uint4*>(bp + 520);
    if (k0 + 32 < Kt) {
      __builtin_prefetch(Ar0 + k0 + 32, 0, 1);
      __builtin_prefetch(bp + bstep, 0, 1);
      __builtin_prefetch(bp + bstep + 512, 0, 1);
    }
#pragma unroll
    for (int i = 0; i < 4; ++i) {
      Frag fa;
      const _Float16* ap = Ar0 + (size_t)i * 16 * Kt + k0 + klo;
      fa.q[0] = *reinterpret_cast<const uint4*>(ap);
      fa.q[1] = *reinterpret_cast<const uint4*>(ap + 8);
      acc[i][0] = __builtin_amdgcn_wmma_f32_16x16x32_f16(false, fa.v, false, fb0.v,
                                                         (short)0, acc[i][0], false, false);
      acc[i][1] = __builtin_amdgcn_wmma_f32_16x16x32_f16(false, fa.v, false, fb1.v,
                                                         (short)0, acc[i][1], false, false);
    }
  }

#pragma unroll
  for (int i = 0; i < 4; ++i) {
#pragma unroll
    for (int t = 0; t < 2; ++t) {
      const int colB = (nt0 + t) * 16 + lm;
      const float bcol = bias ? bias[colB] : 0.f;
#pragma unroll
      for (int r = 0; r < 8; ++r) {
        int row = row0 + i * 16 + r + hs * 8;
        float v = acc[i][t][r] + bcol;
        if (act) v = LEAKY01(v);
        size_t o = (size_t)row * N + colB;
        if (outF) outF[o] = v;
        if (outH) outH[o] = (_Float16)v;
      }
    }
  }
}

// --------------------------- elementwise helpers ---------------------------
__global__ void leaky_k(float* __restrict__ x, int n) {
  int i = blockIdx.x * blockDim.x + threadIdx.x;
  if (i < n) { float v = x[i]; x[i] = LEAKY01(v); }
}

__global__ void reduce_sum_k(const float* __restrict__ x, int n, float* __restrict__ out) {
  __shared__ float sm[256];
  float s = 0.f;
  for (int i = threadIdx.x; i < n; i += 256) s += x[i];
  sm[threadIdx.x] = s; __syncthreads();
  for (int o = 128; o > 0; o >>= 1) {
    if (threadIdx.x < (unsigned)o) sm[threadIdx.x] += sm[threadIdx.x + o];
    __syncthreads();
  }
  if (threadIdx.x == 0) out[0] = sm[0];
}

// ------------------------------- GATv2 core --------------------------------
__global__ void att_init_k(float* __restrict__ out, const float* __restrict__ bias,
                           float* __restrict__ maxb, float* __restrict__ den, int n, int HC) {
  int i = blockIdx.x * blockDim.x + threadIdx.x;
  if (i < n * HC) out[i] = bias[i % HC];
  if (i < n * 8) { maxb[i] = -__builtin_inff(); den[i] = 0.f; }
}

__global__ void gat_logits_k(const float* __restrict__ xl, const float* __restrict__ xr,
                             const int* __restrict__ ei, const float* __restrict__ ea,
                             const float* __restrict__ meansum, float invE,
                             const float* __restrict__ we, const float* __restrict__ attw,
                             float* __restrict__ logits, float* __restrict__ maxb,
                             int E, int n, int HC, int ch)
{
  int idx = blockIdx.x * blockDim.x + threadIdx.x;
  if (idx >= (E + n) * 8) return;
  int h = idx & 7, e = idx >> 3;
  int s, d; float eav;
  if (e < E) { s = ei[e]; d = ei[E + e]; eav = ea[e]; }
  else       { s = d = e - E;            eav = meansum[0] * invE; }
  const float* xls = xl + (size_t)s * HC + h * ch;
  const float* xrd = xr + (size_t)d * HC + h * ch;
  const float* wec = we + h * ch;
  const float* ath = attw + h * ch;
  float acc = 0.f;
  for (int c = 0; c < ch; ++c) {
    float z = xls[c] + xrd[c] + eav * wec[c];
    z = z > 0.f ? z : 0.2f * z;            // leaky_relu 0.2
    acc += ath[c] * z;
  }
  logits[idx] = acc;
  atomicMaxFloat(&maxb[d * 8 + h], acc);
}

__global__ void gat_exp_k(float* __restrict__ logits, const float* __restrict__ maxb,
                          float* __restrict__ den, const int* __restrict__ ei, int E, int n)
{
  int idx = blockIdx.x * blockDim.x + threadIdx.x;
  if (idx >= (E + n) * 8) return;
  int h = idx & 7, e = idx >> 3;
  int d = (e < E) ? ei[E + e] : (e - E);
  float ex = expf(logits[idx] - maxb[d * 8 + h]);
  logits[idx] = ex;
  atomicAdd(&den[d * 8 + h], ex);
}

__global__ void gat_aggr_k(const float* __restrict__ exv, const float* __restrict__ den,
                           const float* __restrict__ xl, const int* __restrict__ ei,
                           float* __restrict__ out, int E, int HC, int chlog2)
{
  int e = blockIdx.x;
  int s, d;
  if (e < E) { s = ei[e]; d = ei[E + e]; } else { s = d = e - E; }
  for (int t = threadIdx.x; t < HC; t += blockDim.x) {
    int h = t >> chlog2;
    float alpha = exv[(size_t)e * 8 + h] / den[d * 8 + h];
    atomicAdd(&out[(size_t)d * HC + t], xl[(size_t)s * HC + t] * alpha);
  }
}

// ------------------------------- batch-norm --------------------------------
// virtual concat of two [rows, split] sources into [rows, cols]
__global__ void bn_stats_k(const float* __restrict__ sA, const float* __restrict__ sB,
                           int rows, int cols, int split,
                           const float* __restrict__ g, const float* __restrict__ bb,
                           float* __restrict__ scale, float* __restrict__ shift)
{
  int c = blockIdx.x * blockDim.x + threadIdx.x;
  if (c >= cols) return;
  const float* src = (c < split) ? (sA + c) : (sB + (c - split));
  float s = 0.f, s2 = 0.f;
  for (int r = 0; r < rows; ++r) { float v = src[(size_t)r * split]; s += v; s2 += v * v; }
  float mu  = s / rows;
  float var = s2 / rows - mu * mu;
  float sc  = g[c] * rsqrtf(var + 1e-5f);
  scale[c] = sc;
  shift[c] = bb[c] - mu * sc;
}

__global__ void bn_apply_f16_k(const float* __restrict__ sA, const float* __restrict__ sB,
                               int rows, int cols, int split,
                               const float* __restrict__ scale, const float* __restrict__ shift,
                               _Float16* __restrict__ out)
{
  int idx = blockIdx.x * blockDim.x + threadIdx.x;
  if (idx >= rows * cols) return;
  int r = idx / cols, c = idx % cols;
  float v = (c < split) ? sA[(size_t)r * split + c] : sB[(size_t)r * split + c - split];
  out[idx] = (_Float16)(v * scale[c] + shift[c]);
}

// --------------------------- routes / nr_routes ----------------------------
__global__ void routes_mean_k(const float* __restrict__ pg, const int* __restrict__ routes,
                              float* __restrict__ pr)
{
  int idx = blockIdx.x * blockDim.x + threadIdx.x;
  if (idx >= 4 * 12 * 1024) return;
  int c  = idx & 1023;
  int r_ = (idx >> 10) % 12;
  int b  = idx / (12 * 1024);
  float acc = 0.f; int cnt = 0;
  for (int nn = 0; nn < 512; ++nn) {
    int gi = b * 512 + nn;
    if (routes[gi] == r_) {
      float v = pg[(size_t)gi * 1024 + c];
      v = LEAKY01(v);                       // leaky applied before routes_mean
      acc += v; cnt++;
    }
  }
  pr[idx] = acc / (float)(cnt > 0 ? cnt : 1);
}

__global__ void nr_routes_k(const float* __restrict__ pr, float* __restrict__ S)
{
  int idx = blockIdx.x * blockDim.x + threadIdx.x;
  if (idx >= 4 * 1024) return;
  int b = idx >> 10;
  int c = idx & 1023;
  float re[12];
  for (int r = 0; r < 12; ++r) re[r] = pr[((size_t)(b * 12 + r)) * 1024 + c];
  float cs[25];
  cs[0] = 1.f;                               // the ones-row cumsum quirk
  float run = 0.f;
  for (int t = 1; t <= 24; ++t) { run += re[(t - 1) % 12]; cs[t] = run; }
  for (int jm = 0; jm < 12; ++jm) {
    int m2 = 11 - jm;
    for (int a = 0; a < 12; ++a) {
      float mv = (cs[a + jm + 1] - cs[a]) / (float)(jm + 1);
      float ot = (cs[a + 1 + m2] - cs[a + 1]) / (float)(m2 > 0 ? m2 : 1);
      size_t base = (((size_t)b * 12 + jm) * 12 + a) * 2048;
      S[base + c]        = mv;
      S[base + 1024 + c] = ot;
    }
  }
}

// fused BN + pairwise expansion into the f16 A-matrix for fc1
__global__ void pair_build_k(const float* __restrict__ S1, const float* __restrict__ S2,
                             const float* __restrict__ scale, const float* __restrict__ shift,
                             _Float16* __restrict__ out)
{
  int row = blockIdx.x;                      // ((b*12+i)*12+j)*12+k
  int k  = row % 12;
  int j  = (row / 12) % 12;
  int ij = row / 144;                        // b*12 + i
  const float* r1 = S1 + ((size_t)ij * 12 + j) * 2048;
  const float* r2 = S2 + ((size_t)ij * 12 + k) * 2048;
  _Float16* o = out + (size_t)row * 4096;
  for (int c = threadIdx.x; c < 4096; c += blockDim.x) {
    float v = (c < 2048) ? r1[c] : r2[c - 2048];
    o[c] = (_Float16)(v * scale[c] + shift[c]);
  }
}

__global__ void head_k(const _Float16* __restrict__ x, const float* __restrict__ w,
                       const float* __restrict__ bb, float* __restrict__ out, int rows)
{
  int i = blockIdx.x * blockDim.x + threadIdx.x;
  if (i >= rows) return;
  const _Float16* xr_ = x + (size_t)i * 64;
  float acc = bb[0];
  for (int c = 0; c < 64; ++c) acc += (float)xr_[c] * w[c];
  out[i]        = 1.f / (1.f + expf(-acc));
  out[rows + i] = (float)(i / 1728);          // ptop_batch
}

// ---------------------------------------------------------------------------
extern "C" void kernel_launch(void* const* d_in, const int* in_sizes, int n_in,
                              void* d_out, int out_size, void* d_ws, size_t ws_size,
                              hipStream_t stream)
{
  (void)in_sizes; (void)n_in; (void)out_size; (void)ws_size;

  const int NG = 2048, F = 16, HC1 = 512, HC3 = 1024;
  const int E1T = 32768, EFT = 65536;
  const int ROWS = 6912, PD = 4096, D2 = 2048;

  const float* p1_x  = (const float*)d_in[0];
  const float* p2_x  = (const float*)d_in[1];
  const float* fg_x  = (const float*)d_in[2];
  const int*   p1_ei = (const int*)d_in[3];
  const int*   p2_ei = (const int*)d_in[4];
  const int*   fg_ei = (const int*)d_in[5];
  const float* p1_ea = (const float*)d_in[6];
  const float* p2_ea = (const float*)d_in[7];
  const float* fg_ea = (const float*)d_in[8];
  const int*   p1_routes = (const int*)d_in[9];
  const int*   p2_routes = (const int*)d_in[10];
  // d_in[11] = epoch (unused)
  const float* g1_wl = (const float*)d_in[12]; const float* g1_bl = (const float*)d_in[13];
  const float* g1_wr = (const float*)d_in[14]; const float* g1_br = (const float*)d_in[15];
  const float* g1_we = (const float*)d_in[16]; const float* g1_att = (const float*)d_in[17];
  const float* g1_bias = (const float*)d_in[18];
  const float* g2_wl = (const float*)d_in[19]; const float* g2_bl = (const float*)d_in[20];
  const float* g2_wr = (const float*)d_in[21]; const float* g2_br = (const float*)d_in[22];
  const float* g2_we = (const float*)d_in[23]; const float* g2_att = (const float*)d_in[24];
  const float* g2_bias = (const float*)d_in[25];
  const float* g3_wl = (const float*)d_in[26]; const float* g3_bl = (const float*)d_in[27];
  const float* g3_wr = (const float*)d_in[28]; const float* g3_br = (const float*)d_in[29];
  const float* g3_we = (const float*)d_in[30]; const float* g3_att = (const float*)d_in[31];
  const float* g3_bias = (const float*)d_in[32];
  const float* bn_both_g = (const float*)d_in[33]; const float* bn_both_b = (const float*)d_in[34];
  const float* bn_ptop_g = (const float*)d_in[35]; const float* bn_ptop_b = (const float*)d_in[36];
  const float* fc_w[6] = { (const float*)d_in[37], (const float*)d_in[39], (const float*)d_in[41],
                           (const float*)d_in[43], (const float*)d_in[45], (const float*)d_in[47] };
  const float* fc_b[6] = { (const float*)d_in[38], (const float*)d_in[40], (const float*)d_in[42],
                           (const float*)d_in[44], (const float*)d_in[46], (const float*)d_in[48] };
  const float* head_w = (const float*)d_in[49];
  const float* head_b = (const float*)d_in[50];

  // ---- workspace bump allocator ----
  char* wsb = (char*)d_ws;
  size_t off = 0;
  auto allocB = [&](size_t bytes) -> void* {
    void* p = wsb + off;
    off += (bytes + 255) & ~(size_t)255;
    return p;
  };
#define AF(n) ((float*)allocB((size_t)(n) * 4))
#define AH(n) ((_Float16*)allocB((size_t)(n) * 2))

  float* xl     = AF((size_t)NG * HC3);
  float* xr     = AF((size_t)NG * HC3);
  float* p1e    = AF((size_t)NG * HC1);
  float* p2e    = AF((size_t)NG * HC1);
  float* fge    = AF((size_t)NG * HC1);
  float* p1g    = AF((size_t)NG * HC3);
  float* p2g    = AF((size_t)NG * HC3);
  float* logits = AF((size_t)(EFT + NG) * 8);
  float* maxb   = AF((size_t)NG * 8);
  float* den    = AF((size_t)NG * 8);
  float* means  = AF(8);
  float* scaleB = AF(PD);
  float* shiftB = AF(PD);
  float* p1r    = AF(4 * 12 * HC3);
  float* p2r    = AF(4 * 12 * HC3);
  float* S1     = AF((size_t)4 * 12 * 12 * D2);
  float* S2     = AF((size_t)4 * 12 * 12 * D2);
  _Float16* xinh = AH((size_t)NG * 32);          // K=16 padded to 32
  _Float16* p1fh = AH((size_t)NG * HC3);
  _Float16* p2fh = AH((size_t)NG * HC3);
  _Float16* Wh   = AH((size_t)4096 * 2048);      // packed-weight slot (max fc1)
  _Float16* P0   = AH((size_t)ROWS * PD);
  _Float16* P1   = AH((size_t)ROWS * D2);
#undef AF
#undef AH

  auto pack = [&](const float* W, int Ktrue, int Kt, int Nn) {
    pack_w_k<<<(Kt * Nn + 255) / 256, 256, 0, stream>>>(W, Wh, Ktrue, Kt, Nn);
  };
  auto gemm = [&](const _Float16* A, const float* bias,
                  float* oF, _Float16* oH, int M, int Nn, int Kt, int act) {
    dim3 g(Nn / 32, M / 256);
    gemm_wmma_k<<<g, 128, 0, stream>>>(A, Wh, bias, oF, oH, M, Nn, Kt, act);
  };
  auto gat = [&](const int* ei, const float* ea, int E, float* meanslot,
                 const float* we, const float* attw, const float* bias,
                 float* out, int HC, int chlog2) {
    int ch = 1 << chlog2;
    int tot8 = (E + NG) * 8;
    att_init_k<<<(NG * HC + 255) / 256, 256, 0, stream>>>(out, bias, maxb, den, NG, HC);
    gat_logits_k<<<(tot8 + 255) / 256, 256, 0, stream>>>(xl, xr, ei, ea, meanslot,
                                                         1.f / (float)E, we, attw,
                                                         logits, maxb, E, NG, HC, ch);
    gat_exp_k<<<(tot8 + 255) / 256, 256, 0, stream>>>(logits, maxb, den, ei, E, NG);
    gat_aggr_k<<<E + NG, 256, 0, stream>>>(logits, den, xl, ei, out, E, HC, chlog2);
  };

  // ---- edge-attr means (one scalar each, reused by GAT1 and GAT3) ----
  reduce_sum_k<<<1, 256, 0, stream>>>(p1_ea, E1T, means + 0);
  reduce_sum_k<<<1, 256, 0, stream>>>(p2_ea, E1T, means + 1);
  reduce_sum_k<<<1, 256, 0, stream>>>(fg_ea, EFT, means + 2);

  // ---- GAT1 on p1 ----
  pad_x_f16_k<<<(NG * 32 + 255) / 256, 256, 0, stream>>>(p1_x, xinh, NG);
  pack(g1_wl, F, 32, HC1);
  gemm(xinh, g1_bl, xl, nullptr, NG, HC1, 32, 0);
  pack(g1_wr, F, 32, HC1);
  gemm(xinh, g1_br, xr, nullptr, NG, HC1, 32, 0);
  gat(p1_ei, p1_ea, E1T, means + 0, g1_we, g1_att, g1_bias, p1e, HC1, 6);
  leaky_k<<<(NG * HC1 + 255) / 256, 256, 0, stream>>>(p1e, NG * HC1);

  // ---- GAT1 on p2 ----
  pad_x_f16_k<<<(NG * 32 + 255) / 256, 256, 0, stream>>>(p2_x, xinh, NG);
  pack(g1_wl, F, 32, HC1);
  gemm(xinh, g1_bl, xl, nullptr, NG, HC1, 32, 0);
  pack(g1_wr, F, 32, HC1);
  gemm(xinh, g1_br, xr, nullptr, NG, HC1, 32, 0);
  gat(p2_ei, p2_ea, E1T, means + 1, g1_we, g1_att, g1_bias, p2e, HC1, 6);
  leaky_k<<<(NG * HC1 + 255) / 256, 256, 0, stream>>>(p2e, NG * HC1);

  // ---- GAT2 on full graph (no leaky) ----
  pad_x_f16_k<<<(NG * 32 + 255) / 256, 256, 0, stream>>>(fg_x, xinh, NG);
  pack(g2_wl, F, 32, HC1);
  gemm(xinh, g2_bl, xl, nullptr, NG, HC1, 32, 0);
  pack(g2_wr, F, 32, HC1);
  gemm(xinh, g2_br, xr, nullptr, NG, HC1, 32, 0);
  gat(fg_ei, fg_ea, EFT, means + 2, g2_we, g2_att, g2_bias, fge, HC1, 6);

  // ---- BN over concat([pXe, fge]) -> f16 inputs for GAT3 ----
  bn_stats_k<<<(2 * HC1 + 255) / 256, 256, 0, stream>>>(p1e, fge, NG, 2 * HC1, HC1,
                                                        bn_both_g, bn_both_b, scaleB, shiftB);
  bn_apply_f16_k<<<(NG * 2 * HC1 + 255) / 256, 256, 0, stream>>>(p1e, fge, NG, 2 * HC1, HC1,
                                                                 scaleB, shiftB, p1fh);
  bn_stats_k<<<(2 * HC1 + 255) / 256, 256, 0, stream>>>(p2e, fge, NG, 2 * HC1, HC1,
                                                        bn_both_g, bn_both_b, scaleB, shiftB);
  bn_apply_f16_k<<<(NG * 2 * HC1 + 255) / 256, 256, 0, stream>>>(p2e, fge, NG, 2 * HC1, HC1,
                                                                 scaleB, shiftB, p2fh);

  // ---- GAT3 on p1 ----
  pack(g3_wl, HC3, HC3, HC3);
  gemm(p1fh, g3_bl, xl, nullptr, NG, HC3, HC3, 0);
  pack(g3_wr, HC3, HC3, HC3);
  gemm(p1fh, g3_br, xr, nullptr, NG, HC3, HC3, 0);
  gat(p1_ei, p1_ea, E1T, means + 0, g3_we, g3_att, g3_bias, p1g, HC3, 7);

  // ---- GAT3 on p2 ----
  pack(g3_wl, HC3, HC3, HC3);
  gemm(p2fh, g3_bl, xl, nullptr, NG, HC3, HC3, 0);
  pack(g3_wr, HC3, HC3, HC3);
  gemm(p2fh, g3_br, xr, nullptr, NG, HC3, HC3, 0);
  gat(p2_ei, p2_ea, E1T, means + 1, g3_we, g3_att, g3_bias, p2g, HC3, 7);

  // ---- route means (leaky fused on read) + nr_routes ----
  routes_mean_k<<<(4 * 12 * HC3 + 255) / 256, 256, 0, stream>>>(p1g, p1_routes, p1r);
  routes_mean_k<<<(4 * 12 * HC3 + 255) / 256, 256, 0, stream>>>(p2g, p2_routes, p2r);
  nr_routes_k<<<(4 * HC3 + 255) / 256, 256, 0, stream>>>(p1r, S1);
  nr_routes_k<<<(4 * HC3 + 255) / 256, 256, 0, stream>>>(p2r, S2);

  // ---- BN over pair (stats from S1/S2: each row repeats exactly R times) ----
  bn_stats_k<<<(PD + 255) / 256, 256, 0, stream>>>(S1, S2, 4 * 12 * 12, PD, D2,
                                                   bn_ptop_g, bn_ptop_b, scaleB, shiftB);
  pair_build_k<<<ROWS, 256, 0, stream>>>(S1, S2, scaleB, shiftB, P0);

  // ---- fc chain (WMMA, leaky, f16 ping-pong) ----
  pack(fc_w[0], 4096, 4096, 2048);
  gemm(P0, fc_b[0], nullptr, P1, ROWS, 2048, 4096, 1);
  pack(fc_w[1], 2048, 2048, 1024);
  gemm(P1, fc_b[1], nullptr, P0, ROWS, 1024, 2048, 1);
  pack(fc_w[2], 1024, 1024, 512);
  gemm(P0, fc_b[2], nullptr, P1, ROWS, 512, 1024, 1);
  pack(fc_w[3], 512, 512, 256);
  gemm(P1, fc_b[3], nullptr, P0, ROWS, 256, 512, 1);
  pack(fc_w[4], 256, 256, 128);
  gemm(P0, fc_b[4], nullptr, P1, ROWS, 128, 256, 1);
  pack(fc_w[5], 128, 128, 64);
  gemm(P1, fc_b[5], nullptr, P0, ROWS, 64, 128, 1);

  // ---- sigmoid head + ptop_batch ----
  head_k<<<(ROWS + 255) / 256, 256, 0, stream>>>(P0, head_w, head_b, (float*)d_out, ROWS);
}
